// Attention_52020643889298
// MI455X (gfx1250) — compile-verified, matrix operand order checked by reading, not run on hardware
//
#include <hip/hip_runtime.h>
#include <stdint.h>

// ---------------------------------------------------------------------------
// Attention block for MI455X (gfx1250): bf16 WMMA GEMMs + flash attention.
// Shapes: B=1, S=2048, D=4096, H=32, KV=8, HD=128.
// GEMMs use double-buffered LDS fed by CDNA5 async global->LDS copies.
// ---------------------------------------------------------------------------

#define S_LEN   2048
#define D_MODEL 4096
#define NH      32
#define NKV     8
#define HD      128
#define SD      (NH * HD)    // 4096
#define KVD     (NKV * HD)   // 1024

typedef __bf16 bf16_t;
typedef __attribute__((ext_vector_type(16))) __bf16 v16bf;
typedef __attribute__((ext_vector_type(8)))  float  v8f;

__device__ __forceinline__ v8f wmma_bf16(v16bf a, v16bf b, v8f c) {
    // D = A(16x32 bf16) * B(32x16 bf16) + C(16x16 f32)
    return __builtin_amdgcn_wmma_f32_16x16x32_bf16(
        /*neg_a=*/false, a, /*neg_b=*/false, b,
        /*c_mod=*/(short)0, c, /*reuse_a=*/false, /*reuse_b=*/false);
}

// CDNA5 async global->LDS copy (ASYNCcnt-tracked, bypasses VGPRs).
// lds: LDS byte offset (low 32 bits of the generic shared address),
// gaddr: 64-bit global address. GV addressing mode (saddr = off).
__device__ __forceinline__ void async_g2lds_b128(uint32_t lds, uint64_t gaddr) {
    asm volatile("global_load_async_to_lds_b128 %0, %1, off"
                 :: "v"(lds), "v"(gaddr) : "memory");
}

__device__ __forceinline__ void wait_async0() {
#if __has_builtin(__builtin_amdgcn_s_wait_asynccnt)
    __builtin_amdgcn_s_wait_asynccnt(0);
#else
    asm volatile("s_wait_asynccnt 0x0" ::: "memory");
#endif
}

__device__ __forceinline__ uint32_t lds_off(const void* p) {
    // Generic shared addresses are {aperture_hi32, lds_offset32}; ISA maps
    // them to LDS by discarding the upper bits.
    return (uint32_t)(uintptr_t)p;
}

// ---------------------------------------------------------------------------
// fp32 -> bf16 straight convert
// ---------------------------------------------------------------------------
__global__ void cvt_f32_bf16(const float* __restrict__ in,
                             bf16_t* __restrict__ out, int n) {
    int i = blockIdx.x * blockDim.x + threadIdx.x;
    if (i < n) out[i] = (bf16_t)in[i];
}

// fp32 [R,C] -> bf16 [C,R] (transpose so WMMA B-fragments are contiguous)
__global__ void cvt_f32_bf16_T(const float* __restrict__ in,
                               bf16_t* __restrict__ out, int R, int C) {
    int i = blockIdx.x * blockDim.x + threadIdx.x;
    if (i >= R * C) return;
    int r = i / C;
    int c = i - r * C;
    out[(size_t)c * R + r] = (bf16_t)in[(size_t)r * C + c];
}

// ---------------------------------------------------------------------------
// RoPE (rotate-half), in place on bf16 activations [S, nheads*HD]
// ---------------------------------------------------------------------------
__global__ void rope_kernel(bf16_t* __restrict__ t, int nheads) {
    int idx = blockIdx.x * blockDim.x + threadIdx.x;
    int total = S_LEN * nheads * (HD / 2);
    if (idx >= total) return;
    int d = idx & 63;                 // 0..63
    int h = (idx >> 6) % nheads;
    int s = idx / (64 * nheads);
    // inv_freq = theta^(-2i/HD) = exp2(-d * log2(10000)/64)
    float invf = exp2f((float)d * -0.20762050593045977f);
    float ang  = (float)s * invf;
    float c, sn;
    __sincosf(ang, &sn, &c);
    bf16_t* base = t + ((size_t)s * nheads + h) * HD;
    float lo = (float)base[d];
    float hi = (float)base[d + 64];
    base[d]      = (bf16_t)(lo * c - hi * sn);
    base[d + 64] = (bf16_t)(hi * c + lo * sn);
}

// ---------------------------------------------------------------------------
// bf16 WMMA GEMM:  C[M,N] = A[M,K] @ Bt[N,K]^T   (Bt is N-major)
// 256 threads = 8 waves; block tile 128x128; each wave: 32x64 (2x4 WMMA tiles)
// Double-buffered LDS, staged with async global->LDS B128 copies.
// ---------------------------------------------------------------------------
__global__ __launch_bounds__(256)
void gemm_bf16_wmma(const bf16_t* __restrict__ A,
                    const bf16_t* __restrict__ Bt,
                    bf16_t* __restrict__ Cb,   // bf16 output (or null)
                    float*  __restrict__ Cf,   // f32 output (or null)
                    int M, int N, int K) {
    __shared__ __align__(32) bf16_t As[2][128 * 32];
    __shared__ __align__(32) bf16_t Bs[2][128 * 32];

    const int tid  = threadIdx.x;
    const int lane = tid & 31;
    const int wid  = tid >> 5;       // 0..7
    const int wm   = wid & 3;        // wave row group: 4 x 32 rows
    const int wn   = wid >> 2;       // wave col group: 2 x 64 cols
    const int ln   = lane & 15;
    const int lh   = lane >> 4;      // K-half select for A/B fragments
    const int m0   = blockIdx.x * 128;
    const int n0   = blockIdx.y * 128;

    // staging assignment: 2 threads per 32-element row, 32B each
    const int srow = tid >> 1;           // 0..127
    const int skk  = (tid & 1) * 16;     // 0 or 16

    v8f acc[2][4] = {};

    // Issue one tile's worth of async copies into buffer `buf`.
    auto stage = [&](int buf, int k0) {
        const uint64_t ga =
            (uint64_t)(uintptr_t)&A[(size_t)(m0 + srow) * K + k0 + skk];
        const uint64_t gb =
            (uint64_t)(uintptr_t)&Bt[(size_t)(n0 + srow) * K + k0 + skk];
        const uint32_t la = lds_off(&As[buf][srow * 32 + skk]);
        const uint32_t lb = lds_off(&Bs[buf][srow * 32 + skk]);
        async_g2lds_b128(la,      ga);
        async_g2lds_b128(la + 16, ga + 16);
        async_g2lds_b128(lb,      gb);
        async_g2lds_b128(lb + 16, gb + 16);
    };

    stage(0, 0);
    wait_async0();
    __syncthreads();

    int buf = 0;
    for (int k0 = 0; k0 < K; k0 += 32) {
        // Kick off the next tile while this one is consumed.
        if (k0 + 32 < K) stage(buf ^ 1, k0 + 32);
        if (k0 + 64 < K) {
            __builtin_prefetch(&A[(size_t)(m0 + srow) * K + k0 + 64 + skk], 0, 3);
            __builtin_prefetch(&Bt[(size_t)(n0 + srow) * K + k0 + 64 + skk], 0, 3);
        }

        v16bf af[2], bfv[4];
#pragma unroll
        for (int i = 0; i < 2; i++)
            af[i] = *(const v16bf*)&As[buf][(wm * 32 + i * 16 + ln) * 32 + lh * 16];
#pragma unroll
        for (int j = 0; j < 4; j++)
            bfv[j] = *(const v16bf*)&Bs[buf][(wn * 64 + j * 16 + ln) * 32 + lh * 16];

#pragma unroll
        for (int i = 0; i < 2; i++)
#pragma unroll
            for (int j = 0; j < 4; j++)
                acc[i][j] = wmma_bf16(af[i], bfv[j], acc[i][j]);

        wait_async0();       // next buffer fully landed in LDS
        __syncthreads();     // everyone done reading current buffer
        buf ^= 1;
    }

    // Epilogue. C layout: lane halves hold M = lh*8 + r, N = ln.
#pragma unroll
    for (int i = 0; i < 2; i++)
#pragma unroll
        for (int j = 0; j < 4; j++) {
            int mb = m0 + wm * 32 + i * 16 + lh * 8;
            int nb = n0 + wn * 64 + j * 16 + ln;
#pragma unroll
            for (int r = 0; r < 8; r++) {
                if (Cf) Cf[(size_t)(mb + r) * N + nb] = acc[i][j][r];
                else    Cb[(size_t)(mb + r) * N + nb] = (bf16_t)acc[i][j][r];
            }
        }
}

// ---------------------------------------------------------------------------
// Flash attention (causal, GQA 4:1).  128 threads = 4 waves.
// Each block: one head, 64 query rows; each wave owns 16 query rows.
// KV streamed in 32-key blocks: S = Q K^T via WMMA, online softmax,
// O += P V via WMMA (V transposed into LDS, P restaged through LDS).
// ---------------------------------------------------------------------------
__global__ __launch_bounds__(128)
void flash_attn(const bf16_t* __restrict__ Q,
                const bf16_t* __restrict__ K,
                const bf16_t* __restrict__ V,
                bf16_t* __restrict__ O) {
    __shared__ __align__(32) bf16_t Vt[128 * 32];       // Vt[d][key]
    __shared__ __align__(32) bf16_t Ps[4][16 * 32];     // per-wave P tile

    const int tid  = threadIdx.x;
    const int lane = tid & 31;
    const int wid  = tid >> 5;
    const int ln   = lane & 15;
    const int lh   = lane >> 4;

    const int qtile = blockIdx.x;          // 0 .. S/64-1
    const int h     = blockIdx.y;          // 0 .. 31
    const int kh    = h >> 2;              // KV head (4 query heads share)
    const int q0    = qtile * 64;
    const int qb    = q0 + wid * 16;       // this wave's first query row

    // Q fragments held in registers for the whole KV stream
    v16bf qf[4];
    {
        const bf16_t* qrow = Q + (size_t)(qb + ln) * SD + h * HD;
#pragma unroll
        for (int c = 0; c < 4; c++)
            qf[c] = *(const v16bf*)(qrow + c * 32 + lh * 16);
    }

    v8f o[8] = {};
    float mi[8], li[8];
#pragma unroll
    for (int r = 0; r < 8; r++) { mi[r] = -1e30f; li[r] = 0.0f; }

    const float scale = 0.08838834764831845f;   // 1/sqrt(128)
    const int nblk = (q0 + 64) / 32;            // causal: keys < q0+64

    for (int jb = 0; jb < nblk; jb++) {
        const int kb = jb * 32;
        __syncthreads();
        // Stage V^T into LDS: thread tid owns dim d = tid, 32 keys.
        {
            const bf16_t* vcol = V + (size_t)kb * KVD + kh * HD + tid;
            bf16_t* dst = &Vt[tid * 32];
#pragma unroll
            for (int kk = 0; kk < 32; kk++)
                dst[kk] = vcol[(size_t)kk * KVD];
        }
        __syncthreads();

        // S = Q K^T for 16 queries x 32 keys (two 16x16 WMMA tiles)
        v8f s0 = {}, s1 = {};
#pragma unroll
        for (int c = 0; c < 4; c++) {
            const bf16_t* k0p = K + (size_t)(kb + ln) * KVD + kh * HD + c * 32 + lh * 16;
            const bf16_t* k1p = K + (size_t)(kb + 16 + ln) * KVD + kh * HD + c * 32 + lh * 16;
            s0 = wmma_bf16(qf[c], *(const v16bf*)k0p, s0);
            s1 = wmma_bf16(qf[c], *(const v16bf*)k1p, s1);
        }

        // Online softmax. Row M = lh*8 + r lives across the 16 lanes of a half.
#pragma unroll
        for (int r = 0; r < 8; r++) {
            const int qi  = qb + lh * 8 + r;
            const int k0i = kb + ln;
            const int k1i = kb + 16 + ln;
            float a = s0[r] * scale; if (k0i > qi) a = -1e30f;
            float b = s1[r] * scale; if (k1i > qi) b = -1e30f;
            float mx = fmaxf(a, b);
#pragma unroll
            for (int off = 1; off < 16; off <<= 1)
                mx = fmaxf(mx, __shfl_xor(mx, off, 32));
            const float mnew  = fmaxf(mi[r], mx);
            const float alpha = __expf(mi[r] - mnew);
            a = __expf(a - mnew);
            b = __expf(b - mnew);
            float rs = a + b;
#pragma unroll
            for (int off = 1; off < 16; off <<= 1)
                rs += __shfl_xor(rs, off, 32);
            li[r] = li[r] * alpha + rs;
            mi[r] = mnew;
#pragma unroll
            for (int dt = 0; dt < 8; dt++) o[dt][r] *= alpha;

            // Restage P (C-fragment layout -> A-fragment layout) via LDS
            const int m = lh * 8 + r;
            Ps[wid][m * 32 + ln]      = (bf16_t)a;
            Ps[wid][m * 32 + 16 + ln] = (bf16_t)b;
        }
        __syncthreads();

        // O += P(16x32) @ V(32x128): 8 WMMA tiles along head_dim
        const v16bf pf = *(const v16bf*)&Ps[wid][ln * 32 + lh * 16];
#pragma unroll
        for (int dt = 0; dt < 8; dt++) {
            const v16bf vf = *(const v16bf*)&Vt[(dt * 16 + ln) * 32 + lh * 16];
            o[dt] = wmma_bf16(pf, vf, o[dt]);
        }
    }

    // Normalize and store O (bf16, [S, H*HD])
#pragma unroll
    for (int dt = 0; dt < 8; dt++) {
#pragma unroll
        for (int r = 0; r < 8; r++) {
            const float val = o[dt][r] / li[r];
            O[(size_t)(qb + lh * 8 + r) * SD + h * HD + dt * 16 + ln] = (bf16_t)val;
        }
    }
}

// ---------------------------------------------------------------------------
// Host-side orchestration
// ---------------------------------------------------------------------------
extern "C" void kernel_launch(void* const* d_in, const int* in_sizes, int n_in,
                              void* d_out, int out_size, void* d_ws, size_t ws_size,
                              hipStream_t stream) {
    (void)in_sizes; (void)n_in; (void)out_size; (void)ws_size;
    const float* x  = (const float*)d_in[0];   // [S, D]
    const float* wq = (const float*)d_in[1];   // [D, SD]
    const float* wk = (const float*)d_in[2];   // [D, KVD]
    const float* wv = (const float*)d_in[3];   // [D, KVD]
    const float* wo = (const float*)d_in[4];   // [SD, D]
    float* out = (float*)d_out;                // [S, D]

    char* ws = (char*)d_ws;
    size_t off = 0;
    auto take = [&](size_t bytes) -> char* {
        char* p = ws + off;
        off = (off + bytes + 255) & ~(size_t)255;
        return p;
    };

    bf16_t* xb  = (bf16_t*)take((size_t)S_LEN * D_MODEL * 2);
    bf16_t* wqt = (bf16_t*)take((size_t)SD  * D_MODEL * 2);
    bf16_t* wkt = (bf16_t*)take((size_t)KVD * D_MODEL * 2);
    bf16_t* wvt = (bf16_t*)take((size_t)KVD * D_MODEL * 2);
    bf16_t* wot = (bf16_t*)take((size_t)D_MODEL * SD * 2);
    bf16_t* Qb  = (bf16_t*)take((size_t)S_LEN * SD  * 2);
    bf16_t* Kb  = (bf16_t*)take((size_t)S_LEN * KVD * 2);
    bf16_t* Vb  = (bf16_t*)take((size_t)S_LEN * KVD * 2);
    bf16_t* Ob  = (bf16_t*)take((size_t)S_LEN * SD  * 2);

    const int T = 256;

    // 1) precision conversion (weights transposed to N-major)
    {
        int n = S_LEN * D_MODEL;
        cvt_f32_bf16<<<(n + T - 1) / T, T, 0, stream>>>(x, xb, n);
        n = D_MODEL * SD;
        cvt_f32_bf16_T<<<(n + T - 1) / T, T, 0, stream>>>(wq, wqt, D_MODEL, SD);
        n = D_MODEL * KVD;
        cvt_f32_bf16_T<<<(n + T - 1) / T, T, 0, stream>>>(wk, wkt, D_MODEL, KVD);
        cvt_f32_bf16_T<<<(n + T - 1) / T, T, 0, stream>>>(wv, wvt, D_MODEL, KVD);
        n = SD * D_MODEL;
        cvt_f32_bf16_T<<<(n + T - 1) / T, T, 0, stream>>>(wo, wot, SD, D_MODEL);
    }

    // 2) QKV projections (bf16 WMMA GEMM)
    {
        dim3 blk(256);
        dim3 gq(S_LEN / 128, SD / 128);
        gemm_bf16_wmma<<<gq, blk, 0, stream>>>(xb, wqt, Qb, nullptr,
                                               S_LEN, SD, D_MODEL);
        dim3 gkv(S_LEN / 128, KVD / 128);
        gemm_bf16_wmma<<<gkv, blk, 0, stream>>>(xb, wkt, Kb, nullptr,
                                                S_LEN, KVD, D_MODEL);
        gemm_bf16_wmma<<<gkv, blk, 0, stream>>>(xb, wvt, Vb, nullptr,
                                                S_LEN, KVD, D_MODEL);
    }

    // 3) RoPE on Q and K
    {
        int nq = S_LEN * NH * (HD / 2);
        rope_kernel<<<(nq + T - 1) / T, T, 0, stream>>>(Qb, NH);
        int nk = S_LEN * NKV * (HD / 2);
        rope_kernel<<<(nk + T - 1) / T, T, 0, stream>>>(Kb, NKV);
    }

    // 4) flash attention
    {
        dim3 grid(S_LEN / 64, NH);
        flash_attn<<<grid, dim3(128), 0, stream>>>(Qb, Kb, Vb, Ob);
    }

    // 5) output projection -> fp32 result
    {
        dim3 blk(256);
        dim3 go(S_LEN / 128, D_MODEL / 128);
        gemm_bf16_wmma<<<go, blk, 0, stream>>>(Ob, wot, nullptr, out,
                                               S_LEN, D_MODEL, SD);
    }
}
